// HybridNet_65481071401691
// MI455X (gfx1250) — matrix operand order repaired and checked by reading, not compiled, not run on hardware
//
#include <hip/hip_runtime.h>

typedef __attribute__((ext_vector_type(8)))  float  v8f;
typedef __attribute__((ext_vector_type(16))) __bf16 v16bf;
typedef __attribute__((ext_vector_type(8)))  __bf16 v8bf;
typedef __attribute__((ext_vector_type(4)))  int    v4i;

#define BATCH 4096
#define FEAT  2048
#define BM 128
#define BN 128
#define BK 64

// ---------------------------------------------------------------------------
// CDNA5 async global->LDS copy (16B per lane), tracked by ASYNCcnt.
// ---------------------------------------------------------------------------
__device__ __forceinline__ void async_copy_b128(void* lds_dst, const void* gsrc) {
#if __has_builtin(__builtin_amdgcn_global_load_async_to_lds_b128)
    __builtin_amdgcn_global_load_async_to_lds_b128(
        (__attribute__((address_space(1))) v4i*)gsrc,
        (__attribute__((address_space(3))) v4i*)lds_dst, 0, 0);
#else
    unsigned lds_off = (unsigned)(unsigned long long)
        (__attribute__((address_space(3))) char*)lds_dst;
    asm volatile("global_load_async_to_lds_b128 %0, %1, off"
                 :: "v"(lds_off), "v"(gsrc) : "memory");
#endif
}

__device__ __forceinline__ void wait_async0() {
#if __has_builtin(__builtin_amdgcn_s_wait_asynccnt)
    __builtin_amdgcn_s_wait_asynccnt(0);
#else
    asm volatile("s_wait_asynccnt 0x0" ::: "memory");
#endif
}

// ---------------------------------------------------------------------------
// 1) Weight prep: W (N x K, fp32 row-major) -> Wt (K x N, bf16 row-major)
//    so each Bt k-row is contiguous in n for the 16B async copies to LDS.
// ---------------------------------------------------------------------------
__global__ __launch_bounds__(256)
void wt_transpose(const float* __restrict__ W0, const float* __restrict__ W1,
                  const float* __restrict__ W2, const float* __restrict__ W3,
                  __bf16* __restrict__ Wt)
{
    __shared__ float tile[32][33];
    const float* Ws[4] = {W0, W1, W2, W3};
    const float* W = Ws[blockIdx.z];
    __bf16* dst = Wt + (size_t)blockIdx.z * FEAT * FEAT;

    const int x = blockIdx.x * 32 + threadIdx.x;   // k
    const int y = blockIdx.y * 32 + threadIdx.y;   // n
#pragma unroll
    for (int j = 0; j < 32; j += 8)
        tile[threadIdx.y + j][threadIdx.x] = W[(size_t)(y + j) * FEAT + x];
    __syncthreads();
    const int tx = blockIdx.y * 32 + threadIdx.x;  // n
    const int ty = blockIdx.x * 32 + threadIdx.y;  // k
#pragma unroll
    for (int j = 0; j < 32; j += 8)
        dst[(size_t)(ty + j) * FEAT + tx] = (__bf16)tile[threadIdx.x][threadIdx.y + j];
}

// ---------------------------------------------------------------------------
// 2) Fused "conv" + sigmoid: h0 = sigmoid(x * sum(conv_w) + conv_b), fp32->bf16
// ---------------------------------------------------------------------------
__global__ __launch_bounds__(256)
void act_sigmoid(const float* __restrict__ x, const float* __restrict__ cw,
                 const float* __restrict__ cb, __bf16* __restrict__ h)
{
    const float s = cw[0] + cw[1] + cw[2] + cw[3];
    const float c = cb[0];
    const size_t i = ((size_t)blockIdx.x * blockDim.x + threadIdx.x) * 8;
    const float4 v0 = *(const float4*)(x + i);
    const float4 v1 = *(const float4*)(x + i + 4);
    float t[8] = {v0.x, v0.y, v0.z, v0.w, v1.x, v1.y, v1.z, v1.w};
    v8bf o;
#pragma unroll
    for (int j = 0; j < 8; ++j) {
        float z = t[j] * s + c;
        o[j] = (__bf16)(1.0f / (1.0f + __expf(-z)));
    }
    *(v8bf*)(h + i) = o;
}

// ---------------------------------------------------------------------------
// 3) Tiled bf16 WMMA GEMM with fused bias + ReLU, bf16 output.
//    C[M,N] = relu(A[M,K] @ Bt[K,N] + bias[N])   (Bt = W^T)
//    Block: 256 threads (8 wave32), tile BM=128 x BN=128, K-step 64.
//    Double-buffered LDS filled by CDNA5 async global->LDS copies (ASYNCcnt),
//    so the tile t+1 copy overlaps the 16 WMMAs of tile t.
// ---------------------------------------------------------------------------
__global__ __launch_bounds__(256)
void gemm_bf16_relu(const __bf16* __restrict__ A, const __bf16* __restrict__ Bt,
                    const float* __restrict__ bias, __bf16* __restrict__ C,
                    int M, int N, int K)
{
    __shared__ __align__(16) __bf16 Ash[2][BM][BK + 8];   // 2 x 128 x 72
    __shared__ __align__(16) __bf16 Bsh[2][BK][BN + 8];   // 2 x 64  x 136

    const int tid  = threadIdx.x;
    const int wave = tid >> 5;
    const int lane = tid & 31;
    const int lh   = lane >> 4;      // half-wave select
    const int lm   = lane & 15;
    const int wm   = wave >> 1;      // 0..3 -> 32-row slice
    const int wn   = wave & 1;       // 0..1 -> 64-col slice
    const int bm0  = blockIdx.y * BM;
    const int bn0  = blockIdx.x * BN;

    v8f acc[2][4];
#pragma unroll
    for (int mi = 0; mi < 2; ++mi)
#pragma unroll
        for (int ni = 0; ni < 4; ++ni)
            acc[mi][ni] = (v8f)(0.0f);

    // async staging indices (each thread issues 8 x b128 = 128B per tile)
    const int ar = tid >> 1;          // 0..127
    const int ac = (tid & 1) * 32;    // 0 / 32
    const int br = tid >> 2;          // 0..63
    const int bc = (tid & 3) * 32;    // 0 / 32 / 64 / 96
    const __bf16* gA = A  + (size_t)(bm0 + ar) * K + ac;   // + k0
    const __bf16* gB = Bt + (size_t)br * N + bn0 + bc;     // + k0*N

    // prologue: fill buffer 0
#pragma unroll
    for (int i = 0; i < 4; ++i) {
        async_copy_b128(&Ash[0][ar][ac + 8 * i], gA + 8 * i);
        async_copy_b128(&Bsh[0][br][bc + 8 * i], gB + 8 * i);
    }
    wait_async0();
    __syncthreads();

    int buf = 0;
    for (int k0 = 0; k0 < K; k0 += BK) {
        // kick off next tile into the other buffer while we compute
        if (k0 + BK < K) {
            const __bf16* nA = gA + (k0 + BK);
            const __bf16* nB = gB + (size_t)(k0 + BK) * N;
#pragma unroll
            for (int i = 0; i < 4; ++i) {
                async_copy_b128(&Ash[buf ^ 1][ar][ac + 8 * i], nA + 8 * i);
                async_copy_b128(&Bsh[buf ^ 1][br][bc + 8 * i], nB + 8 * i);
            }
        }

#pragma unroll
        for (int kk = 0; kk < BK; kk += 32) {
            // A fragments: 16x32 layout — lane m holds K = lh*8..+7 and 16+lh*8..+7
            v16bf afrag[2];
#pragma unroll
            for (int mi = 0; mi < 2; ++mi) {
                const __bf16* ap = &Ash[buf][wm * 32 + mi * 16 + lm][kk];
                v8bf alo = *(const v8bf*)(ap + lh * 8);
                v8bf ahi = *(const v8bf*)(ap + 16 + lh * 8);
                afrag[mi] = __builtin_shufflevector(alo, ahi,
                    0, 1, 2, 3, 4, 5, 6, 7, 8, 9, 10, 11, 12, 13, 14, 15);
            }
#pragma unroll
            for (int ni = 0; ni < 4; ++ni) {
                // B fragment: 32x16 layout — lane n = lm holds K = lh*8..+7 and 16+lh*8..+7
                const __bf16* bp = &Bsh[buf][kk + lh * 8][wn * 64 + ni * 16 + lm];
                v16bf bfrag;
#pragma unroll
                for (int s = 0; s < 8; ++s) {
                    bfrag[s]     = bp[s * (BN + 8)];
                    bfrag[8 + s] = bp[(16 + s) * (BN + 8)];
                }
#pragma unroll
                for (int mi = 0; mi < 2; ++mi)
                    acc[mi][ni] = __builtin_amdgcn_wmma_f32_16x16x32_bf16(
                        false, afrag[mi], false, bfrag,
                        (short)0, acc[mi][ni], false, false);
            }
        }

        wait_async0();      // my async copies into buf^1 have landed
        __syncthreads();    // all waves done computing on buf + copies visible
        buf ^= 1;
    }

    // Epilogue: C/D layout — VGPR v holds M = v + 8*lh, N = lm
#pragma unroll
    for (int ni = 0; ni < 4; ++ni) {
        const int gn = bn0 + wn * 64 + ni * 16 + lm;
        const float bv = bias[gn];
#pragma unroll
        for (int mi = 0; mi < 2; ++mi) {
            const int gm0 = bm0 + wm * 32 + mi * 16 + lh * 8;
#pragma unroll
            for (int v = 0; v < 8; ++v) {
                float r = acc[mi][ni][v] + bv;
                r = r > 0.0f ? r : 0.0f;
                C[(size_t)(gm0 + v) * N + gn] = (__bf16)r;
            }
        }
    }
}

// ---------------------------------------------------------------------------
// 4) Head: out[b, j] = sum_k h[b,k]*Wh[j,k] + bh[j], j in {0,1}, fp32 out.
//    One wave32 per row, shuffle reduction.
// ---------------------------------------------------------------------------
__global__ __launch_bounds__(256)
void head_kernel(const __bf16* __restrict__ H, const float* __restrict__ Wh,
                 const float* __restrict__ bh, float* __restrict__ out, int K)
{
    const int row  = blockIdx.x * 8 + (threadIdx.x >> 5);
    const int lane = threadIdx.x & 31;
    const __bf16* hr = H + (size_t)row * K;
    float s0 = 0.0f, s1 = 0.0f;
    for (int k = lane; k < K; k += 32) {
        const float hv = (float)hr[k];
        s0 += hv * Wh[k];
        s1 += hv * Wh[K + k];
    }
#pragma unroll
    for (int off = 16; off > 0; off >>= 1) {
        s0 += __shfl_xor(s0, off, 32);
        s1 += __shfl_xor(s1, off, 32);
    }
    if (lane == 0) {
        out[row * 2 + 0] = s0 + bh[0];
        out[row * 2 + 1] = s1 + bh[1];
    }
}

// ---------------------------------------------------------------------------
extern "C" void kernel_launch(void* const* d_in, const int* in_sizes, int n_in,
                              void* d_out, int out_size, void* d_ws, size_t ws_size,
                              hipStream_t stream) {
    const float* x      = (const float*)d_in[0];
    const float* conv_w = (const float*)d_in[1];
    const float* conv_b = (const float*)d_in[2];
    const float* W[4]   = {(const float*)d_in[3], (const float*)d_in[5],
                           (const float*)d_in[7], (const float*)d_in[9]};
    const float* b[4]   = {(const float*)d_in[4], (const float*)d_in[6],
                           (const float*)d_in[8], (const float*)d_in[10]};
    const float* Wh     = (const float*)d_in[11];
    const float* bh     = (const float*)d_in[12];
    float* out          = (float*)d_out;

    // Workspace layout: [ 4 x Wt(bf16, K x N) | h0 | h1 ]
    char* ws = (char*)d_ws;
    __bf16* Wt = (__bf16*)ws;
    const size_t wmat = (size_t)FEAT * FEAT;                     // elems / matrix
    __bf16* h0 = (__bf16*)(ws + 4 * wmat * sizeof(__bf16));
    __bf16* h1 = h0 + (size_t)BATCH * FEAT;

    // 1) weight convert + transpose (fp32 NxK -> bf16 KxN), all 4 layers
    wt_transpose<<<dim3(FEAT / 32, FEAT / 32, 4), dim3(32, 8), 0, stream>>>(
        W[0], W[1], W[2], W[3], Wt);

    // 2) fused conv-collapse + sigmoid -> bf16 activations
    act_sigmoid<<<(BATCH * FEAT) / (256 * 8), 256, 0, stream>>>(x, conv_w, conv_b, h0);

    // 3) four WMMA GEMM layers, ping-pong h0 <-> h1
    const dim3 ggrid(FEAT / BN, BATCH / BM);
    gemm_bf16_relu<<<ggrid, 256, 0, stream>>>(h0, Wt + 0 * wmat, b[0], h1, BATCH, FEAT, FEAT);
    gemm_bf16_relu<<<ggrid, 256, 0, stream>>>(h1, Wt + 1 * wmat, b[1], h0, BATCH, FEAT, FEAT);
    gemm_bf16_relu<<<ggrid, 256, 0, stream>>>(h0, Wt + 2 * wmat, b[2], h1, BATCH, FEAT, FEAT);
    gemm_bf16_relu<<<ggrid, 256, 0, stream>>>(h1, Wt + 3 * wmat, b[3], h0, BATCH, FEAT, FEAT);

    // 4) 2048 -> 2 head in fp32
    head_kernel<<<BATCH / 8, 256, 0, stream>>>(h0, Wh, bh, out, FEAT);
}
